// DiTBlock_57887569215897
// MI455X (gfx1250) — compile-verified, matrix operand order checked by reading, not run on hardware
//
#include <hip/hip_runtime.h>
#include <hip/hip_bf16.h>

// ---------------------------------------------------------------------------
// CDNA5 (gfx1250) DiT block: bf16 WMMA GEMMs + flash attention, wave32.
// Double-buffered LDS pipelines; DPP row reductions; async global->LDS copies.
// ---------------------------------------------------------------------------

typedef __bf16 bf16_t;
typedef __attribute__((ext_vector_type(16))) __bf16       v16bf;
typedef __attribute__((ext_vector_type(8)))  float        v8f;
typedef __attribute__((ext_vector_type(4)))  unsigned int u32x4;
typedef __attribute__((ext_vector_type(4)))  int          i32x4;
typedef __attribute__((ext_vector_type(4)))  float        f32x4;

#define WMMA_BF16(A, B, C) \
  __builtin_amdgcn_wmma_f32_16x16x32_bf16(false, (A), false, (B), (short)0, (C), false, false)

#if defined(__AMDGCN__) && __has_builtin(__builtin_amdgcn_global_load_async_to_lds_b128)
#define HAVE_ASYNC_LDS 1
#else
#define HAVE_ASYNC_LDS 0
#endif

#if HAVE_ASYNC_LDS
// Param 0: AS1 (global/__device__) int4*  (probe-confirmed by clang diagnostic)
// Param 1: AS3 (LDS) int4*;  params 2/3: imm offset, imm cpol.
#define ASYNC_COPY_B128(gp, lp)                                        \
  __builtin_amdgcn_global_load_async_to_lds_b128(                      \
      (__attribute__((address_space(1))) i32x4*)(gp),                  \
      (__attribute__((address_space(3))) i32x4*)(lp), 0, 0)
#if __has_builtin(__builtin_amdgcn_s_wait_asynccnt)
#define WAIT_ASYNC() __builtin_amdgcn_s_wait_asynccnt(0)
#else
#define WAIT_ASYNC() asm volatile("s_wait_asynccnt 0" ::: "memory")
#endif
#endif

// Load one 16x32 bf16 WMMA fragment (A layout, or B layout for column-major
// staged data). rowPtr points at K=0 of a 16B-aligned row; kb = lane<16?0:8.
// Lanes 0-15 get K {0..7,16..23}; lanes 16-31 get K {8..15,24..31}
// (CDNA5 ISA 7.12.2 16-bit A/B striping).
__device__ __forceinline__ v16bf ld_frag(const bf16_t* rowPtr, int kb) {
  union { u32x4 u[2]; v16bf v; } cv;
  cv.u[0] = *reinterpret_cast<const u32x4*>(rowPtr + kb);
  cv.u[1] = *reinterpret_cast<const u32x4*>(rowPtr + kb + 16);
  return cv.v;
}

// ---- 16-lane row reductions via DPP (no LDS round-trips) ------------------
#if defined(__AMDGCN__) && __has_builtin(__builtin_amdgcn_update_dpp)
template <int CTRL>
__device__ __forceinline__ float dpp_bcast(float x) {
  return __int_as_float(
      __builtin_amdgcn_update_dpp(0, __float_as_int(x), CTRL, 0xF, 0xF, true));
}
__device__ __forceinline__ float row_max16(float x) {
  x = fmaxf(x, dpp_bcast<0xB1>(x));   // quad_perm(1,0,3,2): xor 1
  x = fmaxf(x, dpp_bcast<0x4E>(x));   // quad_perm(2,3,0,1): xor 2
  x = fmaxf(x, dpp_bcast<0x141>(x));  // row_half_mirror  : xor 4 (quads uniform)
  x = fmaxf(x, dpp_bcast<0x140>(x));  // row_mirror       : xor 8 (octs uniform)
  return x;
}
__device__ __forceinline__ float row_sum16(float x) {
  x += dpp_bcast<0xB1>(x);
  x += dpp_bcast<0x4E>(x);
  x += dpp_bcast<0x141>(x);
  x += dpp_bcast<0x140>(x);
  return x;
}
#else
__device__ __forceinline__ float row_max16(float x) {
  for (int m = 1; m <= 8; m <<= 1) x = fmaxf(x, __shfl_xor(x, m, 32));
  return x;
}
__device__ __forceinline__ float row_sum16(float x) {
  for (int m = 1; m <= 8; m <<= 1) x += __shfl_xor(x, m, 32);
  return x;
}
#endif

__device__ __forceinline__ float gelu_tanh(float x) {
  float x3 = x * x * x;
  return 0.5f * x * (1.0f + tanhf(0.7978845608028654f * (x + 0.044715f * x3)));
}

// ---------------------------------------------------------------------------
// 1) mod = silu(c) @ ada_w + ada_b      (4 x 9216, K = 1024)
// ---------------------------------------------------------------------------
__global__ __launch_bounds__(256)
void ada_kernel(const float* __restrict__ c, const float* __restrict__ ada_w,
                const float* __restrict__ ada_b, float* __restrict__ modout,
                int D, int N9) {
  __shared__ float cs[4][1024];  // silu(c), 16 KB
  const int tid = threadIdx.x;
  for (int r = 0; r < 4; ++r)
    for (int i = tid; i < D; i += 256) {
      float v = c[r * D + i];
      cs[r][i] = v / (1.0f + __expf(-v));  // silu
    }
  __syncthreads();
  const int n = blockIdx.x * 256 + tid;
  float a0 = 0.f, a1 = 0.f, a2 = 0.f, a3 = 0.f;
  for (int k = 0; k < D; ++k) {
    float w = ada_w[(size_t)k * N9 + n];
    a0 += cs[0][k] * w;
    a1 += cs[1][k] * w;
    a2 += cs[2][k] * w;
    a3 += cs[3][k] * w;
  }
  float bb = ada_b[n];
  modout[(size_t)0 * N9 + n] = a0 + bb;
  modout[(size_t)1 * N9 + n] = a1 + bb;
  modout[(size_t)2 * N9 + n] = a2 + bb;
  modout[(size_t)3 * N9 + n] = a3 + bb;
}

// ---------------------------------------------------------------------------
// 2) y = LN(x) * (1 + sc) + sh, cast to bf16.  One workgroup per row, D=1024.
// ---------------------------------------------------------------------------
__global__ __launch_bounds__(256)
void lnmod_kernel(const float* __restrict__ X, const float* __restrict__ modbuf,
                  int sh_chunk, int sc_chunk, bf16_t* __restrict__ Y,
                  int D, int T, int halfT) {
  const int row = blockIdx.x;
  const int b = row / T;
  const int t = row - b * T;
  const int s = (t >= halfT) ? 1 : 0;
  const float* xr = X + (size_t)row * D;
  const int tid = threadIdx.x;

  __shared__ float rbuf[256];
  float vals[4];
  float sum = 0.f;
#pragma unroll
  for (int i = 0; i < 4; ++i) {
    vals[i] = xr[tid + i * 256];
    sum += vals[i];
  }
  rbuf[tid] = sum;
  __syncthreads();
  for (int o = 128; o > 0; o >>= 1) {
    if (tid < o) rbuf[tid] += rbuf[tid + o];
    __syncthreads();
  }
  const float mu = rbuf[0] / (float)D;
  __syncthreads();
  float sq = 0.f;
#pragma unroll
  for (int i = 0; i < 4; ++i) {
    float d = vals[i] - mu;
    sq += d * d;
  }
  rbuf[tid] = sq;
  __syncthreads();
  for (int o = 128; o > 0; o >>= 1) {
    if (tid < o) rbuf[tid] += rbuf[tid + o];
    __syncthreads();
  }
  const float rstd = rsqrtf(rbuf[0] / (float)D + 1e-6f);

  const float* shv = modbuf + ((size_t)(b * 2 + s) * 9 + sh_chunk) * D;
  const float* scv = modbuf + ((size_t)(b * 2 + s) * 9 + sc_chunk) * D;
  bf16_t* yr = Y + (size_t)row * D;
#pragma unroll
  for (int i = 0; i < 4; ++i) {
    int col = tid + i * 256;
    float y = (vals[i] - mu) * rstd;
    y = y * (1.0f + scv[col]) + shv[col];
    yr[col] = (bf16_t)y;
  }
}

// ---------------------------------------------------------------------------
// 3) GEMM:  out = epilogue( A[M,K](bf16) @ W[K,N](f32) + bias )
//    128x128x32 tiles, 8 waves, 32x64 per wave, double-buffered LDS.
//    MODE 1: bf16 out;  MODE 2: gelu -> bf16;  MODE 3: residual + gate*(.)->f32
// ---------------------------------------------------------------------------
constexpr int TM = 128, TN = 128, TK = 32, KPAD = 40;  // 80B rows: 16B aligned

template <int MODE>
__global__ __launch_bounds__(256)
void gemm_bf16_kernel(const bf16_t* __restrict__ A, const float* __restrict__ W,
                      const float* __restrict__ bias, void* __restrict__ outp,
                      const float* __restrict__ residual,
                      const float* __restrict__ modbuf, int gate_chunk,
                      int M, int N, int K, int T, int halfT, int D) {
  __shared__ bf16_t As[2][TM][KPAD];  // activations, row-major
  __shared__ bf16_t Bs[2][TN][KPAD];  // weights staged N-major (transposed)

  const int tid = threadIdx.x;
  const int lane = tid & 31;
  const int wave = tid >> 5;
  const int l16 = lane & 15;
  const int kb = (lane < 16) ? 0 : 8;
  const int rowoff = (lane < 16) ? 0 : 8;
  const int m0 = blockIdx.y * TM;
  const int n0 = blockIdx.x * TN;
  const int wm = (wave & 3) * 32;
  const int wn = (wave >> 2) * 64;

  // Per-thread staging coordinates.
  int ar[2], akc[2];
#pragma unroll
  for (int i = 0; i < 2; ++i) {
    int cidx = tid + i * 256;
    ar[i] = cidx >> 2;
    akc[i] = (cidx & 3) * 8;
  }
  int bkr[4], bnc[4];
#pragma unroll
  for (int i = 0; i < 4; ++i) {
    int cidx = tid + i * 256;
    bkr[i] = cidx >> 5;
    bnc[i] = (cidx & 31) * 4;
  }

  const v8f vzero = {0.f, 0.f, 0.f, 0.f, 0.f, 0.f, 0.f, 0.f};
  v8f acc[2][4];
#pragma unroll
  for (int i = 0; i < 2; ++i)
#pragma unroll
    for (int j = 0; j < 4; ++j) acc[i][j] = vzero;

  const int nt = K / TK;
  u32x4 aReg[2];
  f32x4 bReg[4];

  // ---- prologue: stage tile 0 into buffer 0 ----
#if HAVE_ASYNC_LDS
#pragma unroll
  for (int i = 0; i < 2; ++i)
    ASYNC_COPY_B128(A + (size_t)(m0 + ar[i]) * K + akc[i], &As[0][ar[i]][akc[i]]);
#else
#pragma unroll
  for (int i = 0; i < 2; ++i)
    *reinterpret_cast<u32x4*>(&As[0][ar[i]][akc[i]]) =
        *reinterpret_cast<const u32x4*>(A + (size_t)(m0 + ar[i]) * K + akc[i]);
#endif
#pragma unroll
  for (int i = 0; i < 4; ++i) {
    f32x4 wv = *reinterpret_cast<const f32x4*>(W + (size_t)bkr[i] * N + (n0 + bnc[i]));
    Bs[0][bnc[i] + 0][bkr[i]] = (bf16_t)wv[0];
    Bs[0][bnc[i] + 1][bkr[i]] = (bf16_t)wv[1];
    Bs[0][bnc[i] + 2][bkr[i]] = (bf16_t)wv[2];
    Bs[0][bnc[i] + 3][bkr[i]] = (bf16_t)wv[3];
  }
#if HAVE_ASYNC_LDS
  WAIT_ASYNC();
#endif
  __syncthreads();

  for (int kt = 0; kt < nt; ++kt) {
    const int cur = kt & 1;
    const int nxt = cur ^ 1;
    const bool more = (kt + 1) < nt;
    const int kn = (kt + 1) * TK;

    if (more) {  // issue next tile's loads early (async A goes straight to LDS)
#if HAVE_ASYNC_LDS
#pragma unroll
      for (int i = 0; i < 2; ++i)
        ASYNC_COPY_B128(A + (size_t)(m0 + ar[i]) * K + kn + akc[i],
                        &As[nxt][ar[i]][akc[i]]);
#else
#pragma unroll
      for (int i = 0; i < 2; ++i)
        aReg[i] = *reinterpret_cast<const u32x4*>(A + (size_t)(m0 + ar[i]) * K + kn + akc[i]);
#endif
#pragma unroll
      for (int i = 0; i < 4; ++i)
        bReg[i] = *reinterpret_cast<const f32x4*>(W + (size_t)(kn + bkr[i]) * N + (n0 + bnc[i]));
      if (kt + 2 < nt)  // pull tile kt+2 weights toward L2
        __builtin_prefetch(W + (size_t)(kn + TK + (tid >> 3)) * N + n0 + (tid & 7) * 16, 0, 0);
    }

    // ---- compute on current buffer ----
    v16bf af[2], bfr[4];
#pragma unroll
    for (int tm = 0; tm < 2; ++tm) af[tm] = ld_frag(&As[cur][wm + tm * 16 + l16][0], kb);
#pragma unroll
    for (int tn = 0; tn < 4; ++tn) bfr[tn] = ld_frag(&Bs[cur][wn + tn * 16 + l16][0], kb);
#pragma unroll
    for (int tm = 0; tm < 2; ++tm)
#pragma unroll
      for (int tn = 0; tn < 4; ++tn)
        acc[tm][tn] = WMMA_BF16(af[tm], bfr[tn], acc[tm][tn]);

    if (more) {  // commit next tile
#if !HAVE_ASYNC_LDS
#pragma unroll
      for (int i = 0; i < 2; ++i)
        *reinterpret_cast<u32x4*>(&As[nxt][ar[i]][akc[i]]) = aReg[i];
#endif
#pragma unroll
      for (int i = 0; i < 4; ++i) {
        Bs[nxt][bnc[i] + 0][bkr[i]] = (bf16_t)bReg[i][0];
        Bs[nxt][bnc[i] + 1][bkr[i]] = (bf16_t)bReg[i][1];
        Bs[nxt][bnc[i] + 2][bkr[i]] = (bf16_t)bReg[i][2];
        Bs[nxt][bnc[i] + 3][bkr[i]] = (bf16_t)bReg[i][3];
      }
#if HAVE_ASYNC_LDS
      WAIT_ASYNC();
#endif
      __syncthreads();
    }
  }

  // Epilogue. C layout: VGPR j -> row (j | j+8), col = lane&15.
#pragma unroll
  for (int tm = 0; tm < 2; ++tm) {
#pragma unroll
    for (int tn = 0; tn < 4; ++tn) {
#pragma unroll
      for (int j = 0; j < 8; ++j) {
        int row = m0 + wm + tm * 16 + rowoff + j;
        int col = n0 + wn + tn * 16 + l16;
        float v = acc[tm][tn][j] + bias[col];
        if (MODE == 1) {
          ((bf16_t*)outp)[(size_t)row * N + col] = (bf16_t)v;
        } else if (MODE == 2) {
          ((bf16_t*)outp)[(size_t)row * N + col] = (bf16_t)gelu_tanh(v);
        } else if (MODE == 3) {
          int bb = row / T;
          int t = row - bb * T;
          int s = (t >= halfT) ? 1 : 0;
          float gate = modbuf[((size_t)(bb * 2 + s) * 9 + gate_chunk) * D + col];
          ((float*)outp)[(size_t)row * N + col] =
              residual[(size_t)row * N + col] + gate * v;
        } else {
          ((float*)outp)[(size_t)row * N + col] = v;
        }
      }
    }
  }
}

// ---------------------------------------------------------------------------
// 4) Flash attention per (b, half, head).  N=1024 keys, hd=64.
//    qkv layout: [B*T][3*D], col = which*D + head*64 + dim.
//    swap=1 => queries of half s attend K/V of half 1-s (cross attention).
// ---------------------------------------------------------------------------
__global__ __launch_bounds__(256)
void attn_kernel(const bf16_t* __restrict__ qkv, bf16_t* __restrict__ out,
                 int T, int halfT, int D, int swap) {
  const int qb = blockIdx.x;       // q block of 128 rows
  const int head = blockIdx.y;     // 0..15
  const int bs = blockIdx.z;       // (b, s)
  const int b = bs >> 1;
  const int s = bs & 1;
  const int ks = s ^ swap;

  const int tid = threadIdx.x;
  const int lane = tid & 31;
  const int wave = tid >> 5;
  const int l16 = lane & 15;
  const int kb = (lane < 16) ? 0 : 8;
  const int rowoff = (lane < 16) ? 0 : 8;

  const size_t ld = 3 * (size_t)D;
  const int qrow0 = b * T + s * halfT + qb * 128 + wave * 16;  // 16 q rows/wave
  const int kvrow0 = b * T + ks * halfT;

  __shared__ bf16_t Ksm[2][32][72];    // K tiles (144B rows, 16B aligned)
  __shared__ bf16_t Vts[2][64][40];    // V tiles transposed [dim][kv]
  __shared__ bf16_t Psm[8][16][40];    // per-wave P transpose scratch

  // Staging coordinates: one b128 K chunk + one b128 V chunk per thread.
  const int sr = tid >> 3;
  const int sdc = (tid & 7) * 8;

  // Q fragments held in registers for the whole KV sweep.
  const bf16_t* qptr = qkv + (size_t)(qrow0 + l16) * ld + head * 64;
  const v16bf qf0 = ld_frag(qptr, kb);       // dims 0..31
  const v16bf qf1 = ld_frag(qptr + 32, kb);  // dims 32..63

  const v8f vzero = {0.f, 0.f, 0.f, 0.f, 0.f, 0.f, 0.f, 0.f};
  v8f accO[4];
#pragma unroll
  for (int tn = 0; tn < 4; ++tn) accO[tn] = vzero;
  float mrow[8], lrow[8];
#pragma unroll
  for (int j = 0; j < 8; ++j) { mrow[j] = -1e30f; lrow[j] = 0.f; }
  const float sc = 0.125f;  // 1/sqrt(64)

  const int nt = halfT / 32;
  u32x4 vReg;
#if !HAVE_ASYNC_LDS
  u32x4 kReg;
#endif

  // ---- prologue: stage KV tile 0 into buffer 0 ----
  {
    const bf16_t* kp = qkv + (size_t)(kvrow0 + sr) * ld + D + head * 64 + sdc;
#if HAVE_ASYNC_LDS
    ASYNC_COPY_B128(kp, &Ksm[0][sr][sdc]);
#else
    *reinterpret_cast<u32x4*>(&Ksm[0][sr][sdc]) = *reinterpret_cast<const u32x4*>(kp);
#endif
    const bf16_t* vp = qkv + (size_t)(kvrow0 + sr) * ld + 2 * D + head * 64 + sdc;
    u32x4 vv = *reinterpret_cast<const u32x4*>(vp);
    const bf16_t* ve = reinterpret_cast<const bf16_t*>(&vv);
#pragma unroll
    for (int q = 0; q < 8; ++q) Vts[0][sdc + q][sr] = ve[q];
  }
#if HAVE_ASYNC_LDS
  WAIT_ASYNC();
#endif
  __syncthreads();

  for (int t = 0; t < nt; ++t) {
    const int cur = t & 1;
    const int nxt = cur ^ 1;
    const bool more = (t + 1) < nt;

    if (more) {  // issue next KV tile loads early
      const size_t nrow = (size_t)(kvrow0 + (t + 1) * 32 + sr) * ld;
#if HAVE_ASYNC_LDS
      ASYNC_COPY_B128(qkv + nrow + D + head * 64 + sdc, &Ksm[nxt][sr][sdc]);
#else
      kReg = *reinterpret_cast<const u32x4*>(qkv + nrow + D + head * 64 + sdc);
#endif
      vReg = *reinterpret_cast<const u32x4*>(qkv + nrow + 2 * D + head * 64 + sdc);
    }

    // S = Q(16x64) @ K^T(64x32): row-major K rows double as K^T B-fragments.
    v8f s0 = vzero, s1 = vzero;
    {
      v16bf kf;
      kf = ld_frag(&Ksm[cur][l16][0], kb);        s0 = WMMA_BF16(qf0, kf, s0);
      kf = ld_frag(&Ksm[cur][l16][32], kb);       s0 = WMMA_BF16(qf1, kf, s0);
      kf = ld_frag(&Ksm[cur][16 + l16][0], kb);   s1 = WMMA_BF16(qf0, kf, s1);
      kf = ld_frag(&Ksm[cur][16 + l16][32], kb);  s1 = WMMA_BF16(qf1, kf, s1);
    }

    // Online softmax; each row owned by one 16-lane DPP row.
    float p0[8], p1[8];
#pragma unroll
    for (int j = 0; j < 8; ++j) {
      float a = s0[j] * sc;
      float c2 = s1[j] * sc;
      float mx = row_max16(fmaxf(a, c2));
      float mnew = fmaxf(mrow[j], mx);
      float corr = __expf(mrow[j] - mnew);
      float e0 = __expf(a - mnew);
      float e1 = __expf(c2 - mnew);
      float rs = row_sum16(e0 + e1);
      lrow[j] = lrow[j] * corr + rs;
      mrow[j] = mnew;
      p0[j] = e0;
      p1[j] = e1;
#pragma unroll
      for (int tn = 0; tn < 4; ++tn) accO[tn][j] *= corr;
    }

    // Transpose P (C layout -> A layout) through wave-private LDS.
#pragma unroll
    for (int j = 0; j < 8; ++j) {
      Psm[wave][rowoff + j][l16] = (bf16_t)p0[j];
      Psm[wave][rowoff + j][16 + l16] = (bf16_t)p1[j];
    }
    asm volatile("s_wait_dscnt 0" ::: "memory");  // wave-local LDS RAW fence
    v16bf pf = ld_frag(&Psm[wave][l16][0], kb);

    // O += P(16x32) @ V(32x64) with V staged transposed.
#pragma unroll
    for (int tn = 0; tn < 4; ++tn) {
      v16bf vf = ld_frag(&Vts[cur][tn * 16 + l16][0], kb);
      accO[tn] = WMMA_BF16(pf, vf, accO[tn]);
    }

    if (more) {  // commit next KV tile
#if !HAVE_ASYNC_LDS
      *reinterpret_cast<u32x4*>(&Ksm[nxt][sr][sdc]) = kReg;
#endif
      const bf16_t* ve = reinterpret_cast<const bf16_t*>(&vReg);
#pragma unroll
      for (int q = 0; q < 8; ++q) Vts[nxt][sdc + q][sr] = ve[q];
#if HAVE_ASYNC_LDS
      WAIT_ASYNC();
#endif
      __syncthreads();
    }
  }

  // Normalize and write bf16 output [B*T][D].
#pragma unroll
  for (int tn = 0; tn < 4; ++tn) {
#pragma unroll
    for (int j = 0; j < 8; ++j) {
      int row = qrow0 + rowoff + j;
      int col = head * 64 + tn * 16 + l16;
      out[(size_t)row * D + col] = (bf16_t)(accO[tn][j] / lrow[j]);
    }
  }
}

// ---------------------------------------------------------------------------
// Launch: full DiT block as a 12-kernel pipeline on one stream.
// ---------------------------------------------------------------------------
extern "C" void kernel_launch(void* const* d_in, const int* in_sizes, int n_in,
                              void* d_out, int out_size, void* d_ws, size_t ws_size,
                              hipStream_t stream) {
  (void)in_sizes; (void)n_in; (void)out_size; (void)ws_size;
  const int B = 2, T = 2048, D = 1024, halfT = 1024, Dh = 4096;
  const int M = B * T;  // 4096 rows total

  const float* x       = (const float*)d_in[0];
  const float* c       = (const float*)d_in[1];
  const float* aqkv_w  = (const float*)d_in[2];
  const float* aqkv_b  = (const float*)d_in[3];
  const float* aproj_w = (const float*)d_in[4];
  const float* aproj_b = (const float*)d_in[5];
  const float* cqkv_w  = (const float*)d_in[6];
  const float* cqkv_b  = (const float*)d_in[7];
  const float* cproj_w = (const float*)d_in[8];
  const float* cproj_b = (const float*)d_in[9];
  const float* fc1_w   = (const float*)d_in[10];
  const float* fc1_b   = (const float*)d_in[11];
  const float* fc2_w   = (const float*)d_in[12];
  const float* fc2_b   = (const float*)d_in[13];
  const float* ada_w   = (const float*)d_in[14];
  const float* ada_b   = (const float*)d_in[15];
  float* out = (float*)d_out;

  // Workspace carve-up (~110 MB).
  char* ws = (char*)d_ws;
  size_t off = 0;
  auto carve = [&](size_t bytes) -> char* {
    char* p = ws + off;
    off += (bytes + 255) & ~(size_t)255;
    return p;
  };
  float*  modb   = (float*) carve(sizeof(float) * 4 * 9 * D);
  bf16_t* xmod   = (bf16_t*)carve(sizeof(bf16_t) * (size_t)M * D);
  bf16_t* qkvb   = (bf16_t*)carve(sizeof(bf16_t) * (size_t)M * 3 * D);
  bf16_t* attnb  = (bf16_t*)carve(sizeof(bf16_t) * (size_t)M * D);
  float*  xA     = (float*) carve(sizeof(float) * (size_t)M * D);
  float*  xB     = (float*) carve(sizeof(float) * (size_t)M * D);
  bf16_t* hidden = (bf16_t*)carve(sizeof(bf16_t) * (size_t)M * Dh);

  const dim3 blk(256);

  // adaLN modulation vectors
  ada_kernel<<<dim3(9 * D / 256), blk, 0, stream>>>(c, ada_w, ada_b, modb, D, 9 * D);

  // ---- self attention (per half, shared weights) ----
  lnmod_kernel<<<dim3(M), blk, 0, stream>>>(x, modb, 0, 1, xmod, D, T, halfT);
  gemm_bf16_kernel<1><<<dim3(3 * D / TN, M / TM), blk, 0, stream>>>(
      xmod, aqkv_w, aqkv_b, qkvb, nullptr, nullptr, 0, M, 3 * D, D, T, halfT, D);
  attn_kernel<<<dim3(halfT / 128, 16, B * 2), blk, 0, stream>>>(qkvb, attnb, T, halfT, D, 0);
  gemm_bf16_kernel<3><<<dim3(D / TN, M / TM), blk, 0, stream>>>(
      attnb, aproj_w, aproj_b, xA, x, modb, 2, M, D, D, T, halfT, D);

  // ---- cross attention (halves swap K/V; residual is ORIGINAL x) ----
  lnmod_kernel<<<dim3(M), blk, 0, stream>>>(xA, modb, 3, 4, xmod, D, T, halfT);
  gemm_bf16_kernel<1><<<dim3(3 * D / TN, M / TM), blk, 0, stream>>>(
      xmod, cqkv_w, cqkv_b, qkvb, nullptr, nullptr, 0, M, 3 * D, D, T, halfT, D);
  attn_kernel<<<dim3(halfT / 128, 16, B * 2), blk, 0, stream>>>(qkvb, attnb, T, halfT, D, 1);
  gemm_bf16_kernel<3><<<dim3(D / TN, M / TM), blk, 0, stream>>>(
      attnb, cproj_w, cproj_b, xB, x, modb, 5, M, D, D, T, halfT, D);

  // ---- MLP ----
  lnmod_kernel<<<dim3(M), blk, 0, stream>>>(xB, modb, 6, 7, xmod, D, T, halfT);
  gemm_bf16_kernel<2><<<dim3(Dh / TN, M / TM), blk, 0, stream>>>(
      xmod, fc1_w, fc1_b, hidden, nullptr, nullptr, 0, M, Dh, D, T, halfT, D);
  gemm_bf16_kernel<3><<<dim3(D / TN, M / TM), blk, 0, stream>>>(
      hidden, fc2_w, fc2_b, out, xB, modb, 8, M, D, Dh, T, halfT, D);
}
